// CrossAttention_Mask_30760555774266
// MI455X (gfx1250) — matrix operand order repaired and checked by reading, not compiled, not run on hardware
//
#include <hip/hip_runtime.h>

// ---------------------------------------------------------------------------
// Causal flash-attention forward for MI455X (gfx1250, wave32, WMMA).
// B=16, L=S=2048, E=D=512, fp32 in/out, bf16 WMMA compute, f32 softmax.
// Round 3: vector bf16 converts via __builtin_convertvector (native cvt if
// the backend has it), batched global loads so multiple B128 loads are in
// flight before the LDS store waits on them.
// ---------------------------------------------------------------------------

typedef __attribute__((ext_vector_type(16))) __bf16 v16bf;
typedef __attribute__((ext_vector_type(4)))  __bf16 v4bf;
typedef __attribute__((ext_vector_type(4)))  float  v4f;
typedef __attribute__((ext_vector_type(8)))  float  v8f;

#define B_   16
#define L_   2048
#define E_   512
#define QSTR 520   // Q LDS row stride (halfwords), mult of 8 -> 16B aligned frags
#define KSTR 136   // K LDS row stride
#define VSTR 40    // V^T LDS row stride
#define PSTR 136   // P LDS row stride

union Frag32 {
  uint4 q[2];
  v16bf v;
};

// ---- f32x4 -> bf16x4 (one packed convert pair if the HW op exists) ---------
__device__ __forceinline__ uint2 pk4bf(float4 f) {
  v4f vf = {f.x, f.y, f.z, f.w};
  v4bf h = __builtin_convertvector(vf, v4bf);
  return __builtin_bit_cast(uint2, h);
}

__device__ __forceinline__ unsigned short f2bf(float f) {
  __bf16 h = (__bf16)f;
  return __builtin_bit_cast(unsigned short, h);
}

// ---- DPP rotate within each 16-lane row (wave32: two independent rows) -----
template <int CTRL>
__device__ __forceinline__ float dpp_rot(float x) {
#if __has_builtin(__builtin_amdgcn_update_dpp)
  int r = __builtin_amdgcn_update_dpp(0, __builtin_bit_cast(int, x),
                                      CTRL, 0xF, 0xF, true);
  return __builtin_bit_cast(float, r);
#else
  return __shfl_xor(x, CTRL & 0xF);
#endif
}

__device__ __forceinline__ float row16_max(float x) {
  x = fmaxf(x, dpp_rot<0x121>(x));   // row_ror:1
  x = fmaxf(x, dpp_rot<0x122>(x));   // row_ror:2
  x = fmaxf(x, dpp_rot<0x124>(x));   // row_ror:4
  x = fmaxf(x, dpp_rot<0x128>(x));   // row_ror:8
  return x;
}

__device__ __forceinline__ float row16_sum(float x) {
  x += dpp_rot<0x121>(x);
  x += dpp_rot<0x122>(x);
  x += dpp_rot<0x124>(x);
  x += dpp_rot<0x128>(x);
  return x;
}

__device__ __forceinline__ v16bf lds_frag(const unsigned short* p0,
                                          const unsigned short* p1) {
  Frag32 f;
  f.q[0] = *(const uint4*)p0;          // ds_load_b128
  f.q[1] = *(const uint4*)p1;          // ds_load_b128
  return f.v;
}

__global__ __launch_bounds__(256)
void fa_causal_fwd(const float* __restrict__ Qp,
                   const float* __restrict__ Kp,
                   const float* __restrict__ Vp,
                   float* __restrict__ Op) {
  __shared__ alignas(16) unsigned short Qs[16 * QSTR];   // Q tile, bf16
  __shared__ alignas(16) unsigned short KV[512 * VSTR];  // K chunk / V^T chunk
  __shared__ alignas(16) unsigned short Ps[16 * PSTR];   // probabilities, bf16
  __shared__ float m_run[16], l_run[16], cfs[16];
  __shared__ float pmax[8 * 16], psum[8 * 16];

  const int tid  = threadIdx.x;
  const int wave = tid >> 5;
  const int lane = tid & 31;
  const int half = lane >> 4;
  const int n16  = lane & 15;

  const int wg = blockIdx.x;
  const int b  = wg >> 7;              // 128 query tiles per batch
  const int q0 = (wg & 127) << 4;      // first query row of this tile

  const int kt = wave << 4;            // this wave's key sub-tile in the block
  const int db = wave << 6;            // this wave's 64 output columns
  const float scale = 0.044194173824159216f;  // 1/sqrt(512)

  // ---- load Q tile (16 x 512 f32 -> bf16) into LDS, init softmax state ----
  {
    const float* Qg = Qp + ((size_t)b * L_ + q0) * E_;
#pragma unroll
    for (int i = 0; i < 2; ++i) {
      float4 tmp[4];
#pragma unroll
      for (int j = 0; j < 4; ++j) {                  // 4 loads in flight
        int idx = tid + ((i * 4 + j) << 8);
        tmp[j] = *(const float4*)(Qg + (size_t)(idx >> 7) * E_ +
                                  ((idx & 127) << 2));
      }
#pragma unroll
      for (int j = 0; j < 4; ++j) {
        int idx = tid + ((i * 4 + j) << 8);
        *(uint2*)&Qs[(idx >> 7) * QSTR + ((idx & 127) << 2)] = pk4bf(tmp[j]);
      }
    }
  }
  if (tid < 16) { m_run[tid] = -3.0e38f; l_run[tid] = 0.0f; }
  __syncthreads();

  v8f acc[4] = {};                     // 16 x 64 f32 output accumulator

  const int nkb = (q0 + 16 + 127) >> 7;           // causal key-block count
  for (int kb = 0; kb < nkb; ++kb) {
    const int kb0 = kb << 7;
    v8f sc = {};                       // 16 x 16 score tile (this wave's keys)

    // ================= score phase: S = Q K^T over E ====================
    for (int ec = 0; ec < 4; ++ec) {   // stream K in 128-wide E chunks
      __syncthreads();                 // KV buffer free
      {
        const float* Kg = Kp + ((size_t)b * L_ + kb0) * E_ + (ec << 7);
#pragma unroll
        for (int i = 0; i < 4; ++i) {
          float4 tmp[4];
#pragma unroll
          for (int j = 0; j < 4; ++j) {              // 4 loads in flight
            int idx = tid + ((i * 4 + j) << 8);      // 128 keys x 32 float4
            const float* src = Kg + (size_t)(idx >> 5) * E_ +
                               ((idx & 31) << 2);
            tmp[j] = *(const float4*)src;
            if (ec < 3) __builtin_prefetch(src + 128, 0, 1);  // next e-chunk
          }
#pragma unroll
          for (int j = 0; j < 4; ++j) {
            int idx = tid + ((i * 4 + j) << 8);
            *(uint2*)&KV[(idx >> 5) * KSTR + ((idx & 31) << 2)] = pk4bf(tmp[j]);
          }
        }
      }
      __syncthreads();
#pragma unroll
      for (int es = 0; es < 4; ++es) { // four K=32 WMMA steps
        const int e0 = (ec << 7) + (es << 5);
        const unsigned short* qa = &Qs[n16 * QSTR + e0 + 8 * half];
        v16bf a = lds_frag(qa, qa + 16);
        const unsigned short* kp = &KV[(kt + n16) * KSTR + (es << 5) + 8 * half];
        v16bf bm = lds_frag(kp, kp + 16);
        sc = __builtin_amdgcn_wmma_f32_16x16x32_bf16(
            false, a, false, bm, (short)0, sc, false, false);
      }
    }

    // ---------------- scale + causal mask -------------------------------
    const bool need_mask = (kb0 + kt + 15) > q0;
#pragma unroll
    for (int r = 0; r < 8; ++r) {
      float s = sc[r] * scale;
      if (need_mask) {
        int row = q0 + r + 8 * half;
        int key = kb0 + kt + n16;
        if (key > row) s = -3.0e38f;
      }
      sc[r] = s;
    }

    // ---------------- per-wave row max over its 16 keys (DPP) -----------
#pragma unroll
    for (int r = 0; r < 8; ++r) {
      float m = row16_max(sc[r]);
      if (n16 == 0) pmax[wave * 16 + r + 8 * half] = m;
    }
    __syncthreads();
    if (tid < 16) {                    // online-softmax max update
      float mo = m_run[tid];
      float mn = mo;
#pragma unroll
      for (int w = 0; w < 8; ++w) mn = fmaxf(mn, pmax[w * 16 + tid]);
      cfs[tid]   = __expf(mo - mn);
      m_run[tid] = mn;
    }
    __syncthreads();

    // ---------------- p = exp(s - m), write P, partial sums -------------
#pragma unroll
    for (int r = 0; r < 8; ++r) {
      float mn = m_run[r + 8 * half];
      float p  = __expf(sc[r] - mn);
      Ps[(r + 8 * half) * PSTR + kt + n16] = f2bf(p);
      float s = row16_sum(p);
      if (n16 == 0) psum[wave * 16 + r + 8 * half] = s;
    }
    __syncthreads();
    if (tid < 16) {                    // running denominator
      float ssum = 0.0f;
#pragma unroll
      for (int w = 0; w < 8; ++w) ssum += psum[w * 16 + tid];
      l_run[tid] = l_run[tid] * cfs[tid] + ssum;
    }

    // ---------------- rescale output accumulators -----------------------
#pragma unroll
    for (int r = 0; r < 8; ++r) {
      float c = cfs[r + 8 * half];
      acc[0][r] *= c; acc[1][r] *= c; acc[2][r] *= c; acc[3][r] *= c;
    }

    // ================= AV phase: O += P V, 32-key chunks =================
    for (int kc = 0; kc < 2; ++kc) {
      __syncthreads();                 // KV buffer free (score reads done)
      {
        const float* Vg = Vp + ((size_t)b * L_ + kb0 + (kc << 5)) * E_;
#pragma unroll
        for (int i = 0; i < 4; ++i) {
          float4 tmp[4];
#pragma unroll
          for (int j = 0; j < 4; ++j) {              // 4 loads in flight
            int idx = tid + ((i * 4 + j) << 8);      // 32 keys x 128 float4
            tmp[j] = *(const float4*)(Vg + (size_t)(idx >> 7) * E_ +
                                      ((idx & 127) << 2));
          }
#pragma unroll
          for (int j = 0; j < 4; ++j) {
            int idx = tid + ((i * 4 + j) << 8);
            int row = idx >> 7;
            int c4  = (idx & 127) << 2;
            uint2 pk = pk4bf(tmp[j]);
            KV[(c4 + 0) * VSTR + row] = (unsigned short)(pk.x & 0xFFFFu);
            KV[(c4 + 1) * VSTR + row] = (unsigned short)(pk.x >> 16);
            KV[(c4 + 2) * VSTR + row] = (unsigned short)(pk.y & 0xFFFFu);
            KV[(c4 + 3) * VSTR + row] = (unsigned short)(pk.y >> 16);
          }
        }
      }
      __syncthreads();
      const unsigned short* pa = &Ps[n16 * PSTR + (kc << 5) + 8 * half];
      v16bf a = lds_frag(pa, pa + 16);
#pragma unroll
      for (int t = 0; t < 4; ++t) {
        const unsigned short* vb = &KV[(db + t * 16 + n16) * VSTR + 8 * half];
        v16bf bm = lds_frag(vb, vb + 16);
        acc[t] = __builtin_amdgcn_wmma_f32_16x16x32_bf16(
            false, a, false, bm, (short)0, acc[t], false, false);
      }
    }
  }

  // ---- normalize by row sums and store O (f32) ----
  __syncthreads();
  float linv[8];
#pragma unroll
  for (int r = 0; r < 8; ++r) linv[r] = 1.0f / l_run[r + 8 * half];

  float* Og = Op + ((size_t)b * L_ + q0) * E_;
#pragma unroll
  for (int r = 0; r < 8; ++r) {
    size_t ro = (size_t)(r + 8 * half) * E_;
    Og[ro + db +  0 + n16] = acc[0][r] * linv[r];
    Og[ro + db + 16 + n16] = acc[1][r] * linv[r];
    Og[ro + db + 32 + n16] = acc[2][r] * linv[r];
    Og[ro + db + 48 + n16] = acc[3][r] * linv[r];
  }
}

extern "C" void kernel_launch(void* const* d_in, const int* in_sizes, int n_in,
                              void* d_out, int out_size, void* d_ws, size_t ws_size,
                              hipStream_t stream) {
  (void)in_sizes; (void)n_in; (void)out_size; (void)d_ws; (void)ws_size;
  const float* Q = (const float*)d_in[0];
  const float* K = (const float*)d_in[1];
  const float* V = (const float*)d_in[2];
  float* O = (float*)d_out;
  // one workgroup per (batch, 16-row query tile): 16 * 128 = 2048 blocks
  fa_causal_fwd<<<dim3(B_ * (L_ / 16)), dim3(256), 0, stream>>>(Q, K, V, O);
}
